// SwinTransformerBlock_292057776641
// MI455X (gfx1250) — matrix operand order repaired and checked by reading, compiled, not run on hardware
//
#include <hip/hip_runtime.h>
#include <hip/hip_bf16.h>
#include <cmath>
#include <cstdint>

// ---------------------------------------------------------------------------
// Swin-3D transformer block for MI455X (gfx1250, wave32, WMMA).
// - All GEMMs on v_wmma_f32_16x16x32_f16 (f16 operands, f32 accumulate).
// - WMMA fragments loaded as 16-byte b128 vector loads: weights kept [O][K],
//   K kept token-major (K-dim contiguous), V stored transposed for P@V.
// - GEMM A-tiles staged into LDS with global_load_async_to_lds_b128 (ASYNCcnt).
// - Attention: online (flash) softmax, boundary handling split into fast/edge
//   paths so the hot loop is branch-free.
// ---------------------------------------------------------------------------

typedef _Float16 h16;
typedef __attribute__((ext_vector_type(16))) _Float16 v16h;
typedef __attribute__((ext_vector_type(8)))  _Float16 v8h;
typedef __attribute__((ext_vector_type(8)))  float    v8f;

constexpr int Cc    = 96;
constexpr int HEADS = 3;
constexpr int HD    = 32;
constexpr int Nn    = 343;   // 7*7*7 tokens per window
constexpr int NP    = 352;   // padded to 22*16
constexpr int RT    = 22;    // row tiles per window
constexpr int BW    = 128;   // B * (28/7)^3 windows
constexpr int NT    = 43904; // total tokens B*D*H*W
constexpr int NTT   = NT / 16;
constexpr int MLP   = 384;
constexpr int NN2   = Nn * Nn;

// Workspace layout (bytes), all 16B-aligned. ~100 MB total.
constexpr size_t OFF_BQ   = 0;                                    // w_qkv f16 [288][96]
constexpr size_t OFF_BP   = OFF_BQ   + (size_t)288 * 96 * 2;      // w_proj f16 [96][96]
constexpr size_t OFF_B1   = OFF_BP   + (size_t)96 * 96 * 2;       // w_fc1 f16 [384][96]
constexpr size_t OFF_B2   = OFF_B1   + (size_t)384 * 96 * 2;      // w_fc2 f16 [96][384]
constexpr size_t OFF_WINH = OFF_B2   + (size_t)96 * 384 * 2;      // LN1 windows f16 [128][352][96]
constexpr size_t SZ_WIN   = (size_t)BW * NP * Cc * 2;
constexpr size_t SZ_HB    = (size_t)BW * HEADS * NP * HD * 2;
constexpr size_t OFF_Q    = OFF_WINH + SZ_WIN;                    // q f16 [128][3][352][32] (scaled)
constexpr size_t OFF_K    = OFF_Q    + SZ_HB;                     // k f16 [128][3][352][32]
constexpr size_t OFF_V    = OFF_K    + SZ_HB;                     // vT f16 [128][3][32][352]
constexpr size_t OFF_ATTN = OFF_V    + SZ_HB;                     // attn f16 [128][352][96]
constexpr size_t OFF_X2   = OFF_ATTN + SZ_WIN;                    // residual f32 [43904][96]
constexpr size_t OFF_H2   = OFF_X2   + (size_t)NT * Cc * 4;       // LN2 f16 [43904][96]
constexpr size_t OFF_A1   = OFF_H2   + (size_t)NT * Cc * 2;       // gelu(fc1) f16 [43904][384]
constexpr size_t OFF_BF   = OFF_A1   + (size_t)NT * MLP * 2;      // bias gather f32 [3][343*343]

// --------------------------- WMMA helpers ----------------------------------

__device__ __forceinline__ v8f wmma32(v16h a, v16h b, v8f c) {
  return __builtin_amdgcn_wmma_f32_16x16x32_f16(false, a, false, b, (short)0, c,
                                                false, false);
}

// A fragment 16x32 f16, elem(m,k) = base[m*strideM + k]. Per-lane halves are
// two contiguous 8-half runs -> two b128 loads (base 16B-aligned, strideM*2%16==0).
__device__ __forceinline__ v16h ldA16(const h16* base, int strideM) {
  const int lane = threadIdx.x & 31;
  const int m  = lane & 15;
  const int hs = (lane >> 4) << 3;  // 0 or 8
  const h16* p = base + m * strideM + hs;
  const v8h lo = *(const v8h*)p;         // k = hs .. hs+7
  const v8h hi = *(const v8h*)(p + 16);  // k = 16+hs .. 16+hs+7
  v16h a;
#pragma unroll
  for (int e = 0; e < 8; ++e) { a[e] = lo[e]; a[8 + e] = hi[e]; }
  return a;
}

// B fragment 32x16 f16, elem(k,n) = base[n*strideN + k] (K contiguous).
// Per lane: n = l%16, k = 8*(l/16) + e and 16 + 8*(l/16) + e -> two b128.
__device__ __forceinline__ v16h ldBc(const h16* base, int strideN) {
  const int lane = threadIdx.x & 31;
  const h16* p = base + (lane & 15) * strideN + ((lane >> 4) << 3);
  const v8h lo = *(const v8h*)p;
  const v8h hi = *(const v8h*)(p + 16);
  v16h b;
#pragma unroll
  for (int e = 0; e < 8; ++e) { b[e] = lo[e]; b[8 + e] = hi[e]; }
  return b;
}

// C/D layout: row = r + 8*(lane/16), col = lane%16.
__device__ __forceinline__ int drow(int r) {
  return r + (((threadIdx.x & 31) >> 4) << 3);
}

__device__ __forceinline__ float wredsum32(float v) {
#pragma unroll
  for (int m = 1; m < 32; m <<= 1) v += __shfl_xor(v, m, 32);
  return v;
}
__device__ __forceinline__ float hredsum16(float v) {
#pragma unroll
  for (int m = 1; m < 16; m <<= 1) v += __shfl_xor(v, m, 32);
  return v;
}
__device__ __forceinline__ float hredmax16(float v) {
#pragma unroll
  for (int m = 1; m < 16; m <<= 1) v = fmaxf(v, __shfl_xor(v, m, 32));
  return v;
}

// ------------------- async global->LDS staging (CDNA5) ----------------------

__device__ __forceinline__ void async_ld_b128(uint32_t lds_off, uint64_t gaddr) {
  asm volatile("global_load_async_to_lds_b128 %0, %1, off"
               :: "v"(lds_off), "v"(gaddr) : "memory");
}
__device__ __forceinline__ void wait_async0() {
  asm volatile("s_wait_asynccnt 0x0" ::: "memory");
}

__device__ __forceinline__ void stage_tile(h16* ldst, const h16* gsrc, int halves) {
  const uint32_t lbase = (uint32_t)(uintptr_t)ldst;
  const uint64_t gbase = (uint64_t)(uintptr_t)gsrc;
  const int chunks = halves >> 3;
  for (int i = threadIdx.x; i < chunks; i += blockDim.x)
    async_ld_b128(lbase + (uint32_t)i * 16u, gbase + (uint64_t)i * 16u);
  wait_async0();
  __syncthreads();
}

// --------------------------- kernels ---------------------------------------

__global__ void k_wcvt(h16* dst, const float* src, int n) {
  int i = blockIdx.x * blockDim.x + threadIdx.x;
  if (i < n) dst[i] = (h16)src[i];
}

// Pre-gather relative-position bias: biasF[h][n*Nn+m] = bias_table[rel[n,m]][h].
__global__ void k_bias(const float* __restrict__ bt, const int* __restrict__ ri,
                       float* __restrict__ biasF) {
  int i = blockIdx.x * blockDim.x + threadIdx.x;
  if (i >= HEADS * NN2) return;
  const int h = i / NN2, nm = i % NN2;
  biasF[i] = bt[ri[nm] * HEADS + h];
}

// LN1 + cyclic shift(-3,-3,-3) + window partition -> f16 [128][352][96].
__global__ void k_ln1(const float* __restrict__ x, const float* __restrict__ g,
                      const float* __restrict__ b, h16* __restrict__ winh) {
  int gw = blockIdx.x * (blockDim.x >> 5) + (threadIdx.x >> 5);
  if (gw >= BW * NP) return;
  const int lane = threadIdx.x & 31;
  const int w = gw / NP, n = gw % NP;
  h16* dst = winh + (size_t)gw * Cc;
  if (n >= Nn) {
    dst[lane] = (h16)0.f; dst[lane + 32] = (h16)0.f; dst[lane + 64] = (h16)0.f;
    return;
  }
  const int bi = w >> 6, rem = w & 63;
  const int wz = rem >> 4, wy = (rem >> 2) & 3, wx = rem & 3;
  const int iz = n / 49, r9 = n % 49, iy = r9 / 7, ix = r9 % 7;
  const int dv = (wz * 7 + iz + 3) % 28;
  const int hv = (wy * 7 + iy + 3) % 28;
  const int wv = (wx * 7 + ix + 3) % 28;
  const float* src = x + (size_t)(((bi * 28 + dv) * 28 + hv) * 28 + wv) * Cc;
  float v0 = src[lane], v1 = src[lane + 32], v2 = src[lane + 64];
  float mu = wredsum32(v0 + v1 + v2) * (1.f / 96.f);
  float d0 = v0 - mu, d1 = v1 - mu, d2 = v2 - mu;
  float var = wredsum32(d0 * d0 + d1 * d1 + d2 * d2) * (1.f / 96.f);
  float inv = rsqrtf(var + 1e-5f);
  dst[lane]      = (h16)(d0 * inv * g[lane]      + b[lane]);
  dst[lane + 32] = (h16)(d1 * inv * g[lane + 32] + b[lane + 32]);
  dst[lane + 64] = (h16)(d2 * inv * g[lane + 64] + b[lane + 64]);
}

// QKV GEMM. Each 16-col output tile is entirely inside one of q/k/v and one
// head (16 | 32 | 96), so sel/head/d-base are tile-uniform -> branch per tile.
__global__ void k_qkv(const h16* __restrict__ winh, const h16* __restrict__ bq,
                      const float* __restrict__ bias, h16* __restrict__ qf,
                      h16* __restrict__ kf, h16* __restrict__ vt) {
  __shared__ alignas(16) h16 As[16 * 96];
  const int w = blockIdx.x / RT, rt = blockIdx.x % RT;
  stage_tile(As, winh + ((size_t)w * NP + rt * 16) * Cc, 16 * 96);
  const int wave = threadIdx.x >> 5, col = threadIdx.x & 15;
  const v16h a0 = ldA16(As + 0, 96);
  const v16h a1 = ldA16(As + 32, 96);
  const v16h a2 = ldA16(As + 64, 96);
  for (int ot = wave; ot < 18; ot += 4) {
    v8f acc = {};
    acc = wmma32(a0, ldBc(bq + (size_t)(ot * 16) * 96 + 0, 96), acc);
    acc = wmma32(a1, ldBc(bq + (size_t)(ot * 16) * 96 + 32, 96), acc);
    acc = wmma32(a2, ldBc(bq + (size_t)(ot * 16) * 96 + 64, 96), acc);
    const int ob  = ot * 16;           // tile-uniform
    const int sel = ob / 96;           // 0=q, 1=k, 2=v
    const int oc  = ob % 96;
    const int hh  = oc / 32;
    const int ddb = (oc & 31) + col;   // d index for this lane
    const float bv = bias[ob + col];
    const size_t hb = (size_t)w * HEADS + hh;
    const int n0 = rt * 16 + drow(0);  // first row for this lane half
    if (sel == 0) {
      h16* dst = qf + ((hb * NP + n0) * HD + ddb);
#pragma unroll
      for (int r = 0; r < 8; ++r)
        dst[r * HD] = (h16)((acc[r] + bv) * 0.17677669529663687f);
    } else if (sel == 1) {
      h16* dst = kf + ((hb * NP + n0) * HD + ddb);
#pragma unroll
      for (int r = 0; r < 8; ++r) dst[r * HD] = (h16)(acc[r] + bv);
    } else {
      h16* dst = vt + ((hb * HD + ddb) * NP + n0);  // transposed for P@V
#pragma unroll
      for (int r = 0; r < 8; ++r) dst[r] = (h16)(acc[r] + bv);
    }
  }
}

// Attention: one block per (window, head), one wave per 16-row stripe.
// Online softmax over 11 column-tile pairs; fast path for interior tiles.
__global__ void k_attn(const h16* __restrict__ qf, const h16* __restrict__ kf,
                       const h16* __restrict__ vt,
                       const float* __restrict__ biasF,
                       const float* __restrict__ mask, h16* __restrict__ attn) {
  __shared__ alignas(16) h16 pls_all[4][16 * 32];
  const int w = blockIdx.x / HEADS, h = blockIdx.x % HEADS;
  const int wave = threadIdx.x >> 5, col = threadIdx.x & 15;
  const size_t hb = (size_t)w * HEADS + h;
  const h16* qb  = qf + hb * NP * HD;
  const h16* kb  = kf + hb * NP * HD;
  const h16* vbt = vt + hb * HD * NP;
  const float* mk = mask + (size_t)(w & 63) * NN2;
  const float* bf = biasF + (size_t)h * NN2;
  h16* pls = pls_all[wave];

  for (int rt = wave; rt < RT; rt += 4) {
    const v16h aq = ldA16(qb + rt * 16 * HD, HD);
    v8f acc0 = {}, acc1 = {};
    float mrow[8], srow[8];
    int rowoff[8];
    bool rowok[8];
#pragma unroll
    for (int r = 0; r < 8; ++r) {
      mrow[r] = -INFINITY; srow[r] = 0.f;
      const int nr = rt * 16 + drow(r);
      rowok[r]  = (nr < Nn);
      rowoff[r] = nr * Nn + col;
    }
    const bool rows_full = (rt * 16 + 15) < Nn;  // rt <= 20

    for (int p = 0; p < 11; ++p) {
      v8f z = {};
      v8f s0 = wmma32(aq, ldBc(kb + (2 * p + 0) * 16 * HD, HD), z);
      v8f s1 = wmma32(aq, ldBc(kb + (2 * p + 1) * 16 * HD, HD), z);
      const int mb = 32 * p;
      if (rows_full && p < 10) {
        // interior tile: all (row, col) valid, branch-free bias+mask add
#pragma unroll
        for (int r = 0; r < 8; ++r) {
          const int e0 = rowoff[r] + mb;
          s0[r] += bf[e0] + mk[e0];
          s1[r] += bf[e0 + 16] + mk[e0 + 16];
        }
      } else {
#pragma unroll
        for (int r = 0; r < 8; ++r) {
          const int e0 = rowoff[r] + mb;
          const int mc0 = mb + col, mc1 = mc0 + 16;
          if (rowok[r] && mc0 < Nn) s0[r] += bf[e0] + mk[e0];
          else s0[r] = -1e30f;
          if (rowok[r] && mc1 < Nn) s1[r] += bf[e0 + 16] + mk[e0 + 16];
          else s1[r] = -1e30f;
        }
      }
#pragma unroll
      for (int r = 0; r < 8; ++r) {
        const float mx   = hredmax16(fmaxf(s0[r], s1[r]));
        const float mnew = fmaxf(mrow[r], mx);
        const float sc   = __expf(mrow[r] - mnew);
        mrow[r] = mnew;
        acc0[r] *= sc; acc1[r] *= sc; srow[r] *= sc;
        const float p0 = __expf(s0[r] - mnew);
        const float p1 = __expf(s1[r] - mnew);
        srow[r] += hredsum16(p0 + p1);
        const int row = drow(r);
        pls[row * 32 + col]      = (h16)p0;
        pls[row * 32 + 16 + col] = (h16)p1;
      }
      __builtin_amdgcn_wave_barrier();  // order DS stores before A reload
      const v16h ap = ldA16(pls, 32);   // P 16x32, D-layout -> A-layout via LDS
      acc0 = wmma32(ap, ldBc(vbt + p * 32, NP), acc0);             // d = 0..15
      acc1 = wmma32(ap, ldBc(vbt + 16 * NP + p * 32, NP), acc1);   // d = 16..31
      __builtin_amdgcn_wave_barrier();
    }
#pragma unroll
    for (int r = 0; r < 8; ++r) {
      const float inv = (srow[r] > 0.f) ? (1.f / srow[r]) : 0.f;
      const int n = rt * 16 + drow(r);
      const size_t o = ((size_t)w * NP + n) * Cc + h * HD;
      attn[o + col]      = (h16)(acc0[r] * inv);
      attn[o + 16 + col] = (h16)(acc1[r] * inv);
    }
  }
}

// Output projection + window reverse + reverse shift + residual -> x2 (f32).
__global__ void k_proj(const h16* __restrict__ attn, const h16* __restrict__ bp,
                       const float* __restrict__ bias, const float* __restrict__ x,
                       float* __restrict__ x2) {
  __shared__ alignas(16) h16 As[16 * 96];
  const int w = blockIdx.x / RT, rt = blockIdx.x % RT;
  stage_tile(As, attn + ((size_t)w * NP + rt * 16) * Cc, 16 * 96);
  const int wave = threadIdx.x >> 5, col = threadIdx.x & 15;
  const int bi = w >> 6, rem = w & 63;
  const int wz = rem >> 4, wy = (rem >> 2) & 3, wx = rem & 3;
  for (int ot = wave; ot < 6; ot += 4) {
    v8f acc = {};
#pragma unroll
    for (int ks = 0; ks < 3; ++ks)
      acc = wmma32(ldA16(As + ks * 32, 96), ldBc(bp + (size_t)(ot * 16) * 96 + ks * 32, 96), acc);
    const int c = ot * 16 + col;
    const float bv = bias[c];
#pragma unroll
    for (int r = 0; r < 8; ++r) {
      const int n = rt * 16 + drow(r);
      if (n < Nn) {
        const int iz = n / 49, r9 = n % 49, iy = r9 / 7, ix = r9 % 7;
        const int dv = (wz * 7 + iz + 3) % 28;
        const int hv = (wy * 7 + iy + 3) % 28;
        const int wv = (wx * 7 + ix + 3) % 28;
        const size_t vr = (size_t)(((bi * 28 + dv) * 28 + hv) * 28 + wv) * Cc;
        x2[vr + c] = x[vr + c] + acc[r] + bv;
      }
    }
  }
}

// LN2 over x2 -> f16 tokens [43904][96].
__global__ void k_ln2(const float* __restrict__ x2, const float* __restrict__ g,
                      const float* __restrict__ b, h16* __restrict__ h2) {
  int gw = blockIdx.x * (blockDim.x >> 5) + (threadIdx.x >> 5);
  if (gw >= NT) return;
  const int lane = threadIdx.x & 31;
  const float* src = x2 + (size_t)gw * Cc;
  float v0 = src[lane], v1 = src[lane + 32], v2 = src[lane + 64];
  float mu = wredsum32(v0 + v1 + v2) * (1.f / 96.f);
  float d0 = v0 - mu, d1 = v1 - mu, d2 = v2 - mu;
  float var = wredsum32(d0 * d0 + d1 * d1 + d2 * d2) * (1.f / 96.f);
  float inv = rsqrtf(var + 1e-5f);
  h16* dst = h2 + (size_t)gw * Cc;
  dst[lane]      = (h16)(d0 * inv * g[lane]      + b[lane]);
  dst[lane + 32] = (h16)(d1 * inv * g[lane + 32] + b[lane + 32]);
  dst[lane + 64] = (h16)(d2 * inv * g[lane + 64] + b[lane + 64]);
}

// fc1 GEMM [16x96]@[96x384] + exact-erf GELU -> f16.
__global__ void k_fc1(const h16* __restrict__ h2, const h16* __restrict__ b1w,
                      const float* __restrict__ bias, h16* __restrict__ a1) {
  __shared__ alignas(16) h16 As[16 * 96];
  const int rt = blockIdx.x;
  stage_tile(As, h2 + (size_t)rt * 16 * Cc, 16 * 96);
  const int wave = threadIdx.x >> 5, col = threadIdx.x & 15;
  for (int ot = wave; ot < 24; ot += 4) {
    v8f acc = {};
#pragma unroll
    for (int ks = 0; ks < 3; ++ks)
      acc = wmma32(ldA16(As + ks * 32, 96), ldBc(b1w + (size_t)(ot * 16) * 96 + ks * 32, 96), acc);
    const int o = ot * 16 + col;
    const float bv = bias[o];
    h16* dst = a1 + (size_t)(rt * 16 + drow(0)) * MLP + o;
#pragma unroll
    for (int r = 0; r < 8; ++r) {
      const float v = acc[r] + bv;
      dst[(size_t)r * MLP] = (h16)(0.5f * v * (1.f + erff(v * 0.7071067811865476f)));
    }
  }
}

// fc2 GEMM [16x384]@[384x96] + bias + residual -> f32 output.
__global__ void k_fc2(const h16* __restrict__ a1, const h16* __restrict__ b2w,
                      const float* __restrict__ bias, const float* __restrict__ x2,
                      float* __restrict__ out) {
  __shared__ alignas(16) h16 As[16 * MLP];
  const int rt = blockIdx.x;
  stage_tile(As, a1 + (size_t)rt * 16 * MLP, 16 * MLP);
  const int wave = threadIdx.x >> 5, col = threadIdx.x & 15;
  for (int ot = wave; ot < 6; ot += 4) {
    v8f acc = {};
#pragma unroll
    for (int ks = 0; ks < 12; ++ks)
      acc = wmma32(ldA16(As + ks * 32, MLP), ldBc(b2w + (size_t)(ot * 16) * MLP + ks * 32, MLP), acc);
    const int o = ot * 16 + col;
    const float bv = bias[o];
    const size_t base = (size_t)(rt * 16 + drow(0)) * Cc + o;
#pragma unroll
    for (int r = 0; r < 8; ++r)
      out[base + (size_t)r * Cc] = acc[r] + bv + x2[base + (size_t)r * Cc];
  }
}

// --------------------------- launch ----------------------------------------

extern "C" void kernel_launch(void* const* d_in, const int* in_sizes, int n_in,
                              void* d_out, int out_size, void* d_ws, size_t ws_size,
                              hipStream_t stream) {
  (void)in_sizes; (void)n_in; (void)out_size; (void)ws_size;
  const float* x          = (const float*)d_in[0];
  const float* mask       = (const float*)d_in[1];
  const float* g1         = (const float*)d_in[2];
  const float* b1         = (const float*)d_in[3];
  const float* w_qkv      = (const float*)d_in[4];
  const float* b_qkv      = (const float*)d_in[5];
  const float* bias_table = (const float*)d_in[6];
  const float* w_proj     = (const float*)d_in[7];
  const float* b_proj     = (const float*)d_in[8];
  const float* g2         = (const float*)d_in[9];
  const float* b2         = (const float*)d_in[10];
  const float* w_fc1      = (const float*)d_in[11];
  const float* b_fc1      = (const float*)d_in[12];
  const float* w_fc2      = (const float*)d_in[13];
  const float* b_fc2      = (const float*)d_in[14];
  const int*   rel_index  = (const int*)d_in[15];
  float* out = (float*)d_out;
  char*  ws  = (char*)d_ws;

  h16* BQ    = (h16*)(ws + OFF_BQ);
  h16* BP    = (h16*)(ws + OFF_BP);
  h16* B1W   = (h16*)(ws + OFF_B1);
  h16* B2W   = (h16*)(ws + OFF_B2);
  h16* WINH  = (h16*)(ws + OFF_WINH);
  h16* QF    = (h16*)(ws + OFF_Q);
  h16* KF    = (h16*)(ws + OFF_K);
  h16* VT    = (h16*)(ws + OFF_V);
  h16* ATTN  = (h16*)(ws + OFF_ATTN);
  float* X2  = (float*)(ws + OFF_X2);
  h16* H2    = (h16*)(ws + OFF_H2);
  h16* A1    = (h16*)(ws + OFF_A1);
  float* BF  = (float*)(ws + OFF_BF);

  k_wcvt<<<(288 * 96 + 255) / 256, 256, 0, stream>>>(BQ, w_qkv, 288 * 96);
  k_wcvt<<<(96 * 96 + 255) / 256, 256, 0, stream>>>(BP, w_proj, 96 * 96);
  k_wcvt<<<(384 * 96 + 255) / 256, 256, 0, stream>>>(B1W, w_fc1, 384 * 96);
  k_wcvt<<<(96 * 384 + 255) / 256, 256, 0, stream>>>(B2W, w_fc2, 96 * 384);
  k_bias<<<(HEADS * NN2 + 255) / 256, 256, 0, stream>>>(bias_table, rel_index, BF);

  k_ln1<<<(BW * NP + 7) / 8, 256, 0, stream>>>(x, g1, b1, WINH);
  k_qkv<<<BW * RT, 128, 0, stream>>>(WINH, BQ, b_qkv, QF, KF, VT);
  k_attn<<<BW * HEADS, 128, 0, stream>>>(QF, KF, VT, BF, mask, ATTN);
  k_proj<<<BW * RT, 128, 0, stream>>>(ATTN, BP, b_proj, x, X2);
  k_ln2<<<(NT + 7) / 8, 256, 0, stream>>>(X2, g2, b2, H2);
  k_fc1<<<NTT, 128, 0, stream>>>(H2, B1W, b_fc1, A1);
  k_fc2<<<NTT, 128, 0, stream>>>(A1, B2W, b_fc2, X2, out);
}
